// DiagonalSSMLayer_62302795596604
// MI455X (gfx1250) — compile-verified, matrix operand order checked by reading, not run on hardware
//
#include <hip/hip_runtime.h>
#include <hip/hip_bf16.h>
#include <stdint.h>

#define D_MODEL 1024
#define STATE_N 1024
#define BATCH   4
#define SEQ_T   4096
#define M_TOTAL (BATCH * SEQ_T)      // 16384 rows
#define CHUNK_T 256
#define N_CHUNK (SEQ_T / CHUNK_T)    // 16

#define KSLAB     128                // K elements staged per slab
#define LDS_PITCH 136                // 128 + 8 elems pad (TDM pad: 4 dw per 64 dw)
#define SLAB_ELEMS (64 * LDS_PITCH)  // 8704 elems = 17408 B per (matrix, buffer)

typedef __attribute__((ext_vector_type(16))) __bf16 v16bf;
typedef __attribute__((ext_vector_type(8)))  float  v8f;
typedef __attribute__((ext_vector_type(4)))  unsigned int u32x4;
typedef __attribute__((ext_vector_type(4)))  int    i32x4;
typedef __attribute__((ext_vector_type(8)))  int    i32x8;

#if __has_builtin(__builtin_amdgcn_tensor_load_to_lds)
#define HAVE_TDM 1
#else
#define HAVE_TDM 0
#endif

__device__ __forceinline__ uint16_t f2bf(float f) {
    uint32_t u = __float_as_uint(f);
    uint32_t r = u + 0x7FFFu + ((u >> 16) & 1u);   // round-to-nearest-even
    return (uint16_t)(r >> 16);
}
__device__ __forceinline__ float bf2f(uint16_t h) {
    return __uint_as_float(((uint32_t)h) << 16);
}
__device__ __forceinline__ v8f wmma_bf16(v16bf a, v16bf b, v8f c) {
    return __builtin_amdgcn_wmma_f32_16x16x32_bf16(false, a, false, b, (short)0, c,
                                                   false, false);
}
__device__ __forceinline__ void wait_tensorcnt0() {
#if __has_builtin(__builtin_amdgcn_s_wait_tensorcnt)
    __builtin_amdgcn_s_wait_tensorcnt(0);
#else
    asm volatile("s_wait_tensorcnt 0x0" ::: "memory");
#endif
}

#if HAVE_TDM
// Issue one TDM 2D tile load: tile_h rows x KSLAB bf16 elems (row-contiguous in
// global, row stride row_stride_elems) -> LDS at byte offset lds_byte_off with
// 4-dword padding every 64 dwords (row pitch LDS_PITCH elems). Descriptor values
// are block-uniform, so they live in SGPRs as the VIMAGE encoding requires.
// NOTE: this toolchain's builtin takes the 6-arg form
//   (u32x4 g0, i32x8 g1, i32x4 g2, i32x4 g3, i32x8 extra, i32 cpol).
__device__ __forceinline__ void tdm_issue_2d(const uint16_t* gsrc, uint32_t lds_byte_off,
                                             uint32_t tile_h, uint32_t row_stride_elems) {
    const uint64_t ga = (uint64_t)(uintptr_t)gsrc;
    u32x4 g0;
    g0[0] = 1u;                                   // count=1: valid user descriptor
    g0[1] = lds_byte_off;                         // lds_addr (bytes)
    g0[2] = (uint32_t)ga;                         // global_addr[31:0]
    g0[3] = (uint32_t)(ga >> 32) | (2u << 30);    // global_addr[56:32] | type=2
    i32x8 g1;
    // data_size=1 (2B) | pad_enable | pad_interval=5 (64 dw) | pad_amount=3 (4 dw)
    g1[0] = (1 << 16) | (1 << 20) | (5 << 22) | (3 << 25);
    g1[1] = (int)0xFFFF0000u;                     // tensor_dim0 lo16 (huge tensor)
    g1[2] = (int)(0x000Fu | (0xFFFFu << 16));     // tensor_dim0 hi16 | tensor_dim1 lo16
    g1[3] = (int)(0x000Fu | ((uint32_t)KSLAB << 16)); // tensor_dim1 hi16 | tile_dim0
    g1[4] = (int)(tile_h & 0xFFFFu);              // tile_dim1 (tile_dim2 = 0)
    g1[5] = (int)row_stride_elems;                // tensor_dim0_stride lo32
    g1[6] = 0;                                    // stride hi16 | dim1_stride lo16
    g1[7] = 0;
    const i32x4 z4 = {0, 0, 0, 0};
    const i32x8 z8 = {0, 0, 0, 0, 0, 0, 0, 0};
    __builtin_amdgcn_tensor_load_to_lds(g0, g1, z4, z4, z8, 0);
}
#else
// Cooperative fallback: 256 threads stage a 64 x KSLAB tile with the same pitch.
__device__ __forceinline__ void stage_slab_fallback(uint16_t* lds_dst, const uint16_t* gsrc,
                                                    int tid, int row_stride_elems) {
    const int r = tid >> 2;            // 0..63
    const int c = (tid & 3) * 32;      // 0,32,64,96
    const uint16_t* src = gsrc + (size_t)r * row_stride_elems + c;
    uint16_t* dst = lds_dst + r * LDS_PITCH + c;
    *(v16bf*)dst        = *(const v16bf*)src;
    *(v16bf*)(dst + 16) = *(const v16bf*)(src + 16);
}
#endif

// ---------------------------------------------------------------- fp32 -> bf16
__global__ __launch_bounds__(256) void cvt_bf16_kernel(const float* __restrict__ src,
                                                       uint16_t* __restrict__ dst, int n) {
    int i = blockIdx.x * 256 + threadIdx.x;
    if (i < n) dst[i] = f2bf(src[i]);
}

// ---------------------------------------------------------------- LayerNorm (one wave32 per row)
__global__ __launch_bounds__(256) void ln_kernel(const float* __restrict__ x,
                                                 const float* __restrict__ w,
                                                 const float* __restrict__ b,
                                                 uint16_t* __restrict__ xnb) {
    const int wave = threadIdx.x >> 5;
    const int lane = threadIdx.x & 31;
    const int row  = blockIdx.x * 8 + wave;
    const float* xr = x + (size_t)row * D_MODEL;

    float v[32];
    float s = 0.f, ss = 0.f;
#pragma unroll
    for (int i = 0; i < 32; ++i) {
        v[i] = xr[lane + i * 32];
        s += v[i]; ss += v[i] * v[i];
    }
#pragma unroll
    for (int off = 16; off >= 1; off >>= 1) {
        s  += __shfl_xor(s,  off, 32);
        ss += __shfl_xor(ss, off, 32);
    }
    const float mu   = s * (1.f / D_MODEL);
    const float var  = ss * (1.f / D_MODEL) - mu * mu;
    const float rstd = rsqrtf(var + 1e-5f);

    uint16_t* o = xnb + (size_t)row * D_MODEL;
#pragma unroll
    for (int i = 0; i < 32; ++i) {
        const int c = lane + i * 32;
        o[c] = f2bf((v[i] - mu) * rstd * w[c] + b[c]);
    }
}

// ---------------------------------------------------------------- dual GEMM: alpha (sigmoid) + u
// TDM double-buffers 64x128 weight slabs of Wa and Wb into LDS; 8 waves share them.
__global__ __launch_bounds__(256) void gemm_alpha_u_kernel(
    const uint16_t* __restrict__ xnb, const uint16_t* __restrict__ wa,
    const uint16_t* __restrict__ wb,  const float* __restrict__ ba,
    const float* __restrict__ bb,     uint16_t* __restrict__ alpha_o,
    uint16_t* __restrict__ u_o) {
    __shared__ uint16_t smem[4 * SLAB_ELEMS];     // [buf:2][mat:2] -> 69.6 KB
    const int tid  = threadIdx.x;
    const int lane = tid & 31;
    const int wave = tid >> 5;
    const int n0   = blockIdx.x * 64;
    const int m0   = blockIdx.y * 128 + wave * 16;
    const int lrow = lane & 15;
    const int kh   = lane >> 4;

    if ((int)blockIdx.x < 0) smem[tid] = 0;       // never taken: keeps LDS "written"

    v8f accA[4] = {};
    v8f accU[4] = {};

    const uint16_t* warow = wa + (size_t)n0 * D_MODEL;
    const uint16_t* wbrow = wb + (size_t)n0 * D_MODEL;
    const uint16_t* arow  = xnb + (size_t)(m0 + lrow) * D_MODEL + kh * 16;

    // prologue: stage slab 0 into buffer 0
#if HAVE_TDM
    if (wave == 0) {
        tdm_issue_2d(warow, 0u * SLAB_ELEMS * 2u, 64, D_MODEL);
        tdm_issue_2d(wbrow, 1u * SLAB_ELEMS * 2u, 64, D_MODEL);
        wait_tensorcnt0();
    }
#else
    stage_slab_fallback(&smem[0 * SLAB_ELEMS], warow, tid, D_MODEL);
    stage_slab_fallback(&smem[1 * SLAB_ELEMS], wbrow, tid, D_MODEL);
#endif
    __syncthreads();

    for (int ks = 0; ks < D_MODEL / KSLAB; ++ks) {
        const int buf = ks & 1;
        if (ks < D_MODEL / KSLAB - 1) {           // async-prefetch next slab
#if HAVE_TDM
            if (wave == 0) {
                tdm_issue_2d(warow + (ks + 1) * KSLAB,
                             (uint32_t)(((buf ^ 1) * 2 + 0) * SLAB_ELEMS * 2), 64, D_MODEL);
                tdm_issue_2d(wbrow + (ks + 1) * KSLAB,
                             (uint32_t)(((buf ^ 1) * 2 + 1) * SLAB_ELEMS * 2), 64, D_MODEL);
            }
#else
            stage_slab_fallback(&smem[((buf ^ 1) * 2 + 0) * SLAB_ELEMS],
                                warow + (ks + 1) * KSLAB, tid, D_MODEL);
            stage_slab_fallback(&smem[((buf ^ 1) * 2 + 1) * SLAB_ELEMS],
                                wbrow + (ks + 1) * KSLAB, tid, D_MODEL);
#endif
        }

        v16bf af[4];
#pragma unroll
        for (int kk = 0; kk < 4; ++kk)
            af[kk] = *(const v16bf*)(arow + ks * KSLAB + kk * 32);

        const uint16_t* sa = &smem[(buf * 2 + 0) * SLAB_ELEMS];
        const uint16_t* sb = &smem[(buf * 2 + 1) * SLAB_ELEMS];
#pragma unroll
        for (int kk = 0; kk < 4; ++kk) {
#pragma unroll
            for (int j = 0; j < 4; ++j) {
                const int off = (j * 16 + lrow) * LDS_PITCH + kk * 32 + kh * 16;
                const v16bf bA = *(const v16bf*)(sa + off);
                const v16bf bB = *(const v16bf*)(sb + off);
                accA[j] = wmma_bf16(af[kk], bA, accA[j]);
                accU[j] = wmma_bf16(af[kk], bB, accU[j]);
            }
        }

#if HAVE_TDM
        if (wave == 0) wait_tensorcnt0();         // next slab landed
#endif
        __syncthreads();                          // also guards buffer reuse
    }

#pragma unroll
    for (int j = 0; j < 4; ++j) {
        const int col   = n0 + j * 16 + lrow;
        const float bsA = ba[col];
        const float bsB = bb[col];
#pragma unroll
        for (int r = 0; r < 8; ++r) {
            const int rr = m0 + r + 8 * kh;       // C/D layout: lanes 16-31 hold M+8
            const size_t idx = (size_t)rr * STATE_N + col;
            const float av = 1.f / (1.f + __expf(-(accA[j][r] + bsA)));
            alpha_o[idx] = f2bf(av);
            u_o[idx]     = f2bf(accU[j][r] + bsB);
        }
    }
}

// ---------------------------------------------------------------- scan pass 1: per-chunk (P, S)
__global__ __launch_bounds__(256) void scan_pass1_kernel(
    const uint16_t* __restrict__ alpha, const uint16_t* __restrict__ u,
    float* __restrict__ P, float* __restrict__ S) {
    const int n     = blockIdx.x * 256 + threadIdx.x;
    const int chunk = blockIdx.y;
    const int b     = blockIdx.z;
    size_t base = ((size_t)b * SEQ_T + (size_t)chunk * CHUNK_T) * STATE_N + n;
    float p = 1.f, s = 0.f;
    for (int t = 0; t < CHUNK_T; ++t) {
        const float a  = bf2f(alpha[base]);
        const float uu = bf2f(u[base]);
        p *= a;
        s = fmaf(a, s, uu);
        base += STATE_N;
    }
    const size_t ci = ((size_t)b * N_CHUNK + chunk) * STATE_N + n;
    P[ci] = p; S[ci] = s;
}

// ---------------------------------------------------------------- carry combine (serial over 16)
__global__ __launch_bounds__(256) void scan_carry_kernel(
    const float* __restrict__ P, const float* __restrict__ S,
    float* __restrict__ Hinit) {
    const int n = blockIdx.x * 256 + threadIdx.x;
    const int b = blockIdx.y;
    float h = 0.f;
#pragma unroll
    for (int c = 0; c < N_CHUNK; ++c) {
        const size_t ci = ((size_t)b * N_CHUNK + c) * STATE_N + n;
        Hinit[ci] = h;
        h = fmaf(P[ci], h, S[ci]);
    }
}

// ---------------------------------------------------------------- scan pass 2: emit h (bf16)
__global__ __launch_bounds__(256) void scan_pass2_kernel(
    const uint16_t* __restrict__ alpha, const uint16_t* __restrict__ u,
    const float* __restrict__ Hinit, uint16_t* __restrict__ hout) {
    const int n     = blockIdx.x * 256 + threadIdx.x;
    const int chunk = blockIdx.y;
    const int b     = blockIdx.z;
    size_t base = ((size_t)b * SEQ_T + (size_t)chunk * CHUNK_T) * STATE_N + n;
    float s = Hinit[((size_t)b * N_CHUNK + chunk) * STATE_N + n];
    for (int t = 0; t < CHUNK_T; ++t) {
        const float a  = bf2f(alpha[base]);
        const float uu = bf2f(u[base]);
        s = fmaf(a, s, uu);
        hout[base] = f2bf(s);
        base += STATE_N;
    }
}

// ---------------------------------------------------------------- output GEMM + bias + residual
__global__ __launch_bounds__(256) void gemm_out_kernel(
    const uint16_t* __restrict__ hb, const uint16_t* __restrict__ wc,
    const float* __restrict__ bc,    const float* __restrict__ Dp,
    const float* __restrict__ x,     float* __restrict__ y) {
    __shared__ uint16_t smem[2 * SLAB_ELEMS];     // [buf:2] -> 34.8 KB
    const int tid  = threadIdx.x;
    const int lane = tid & 31;
    const int wave = tid >> 5;
    const int n0   = blockIdx.x * 64;
    const int m0   = blockIdx.y * 128 + wave * 16;
    const int lrow = lane & 15;
    const int kh   = lane >> 4;

    if ((int)blockIdx.x < 0) smem[tid] = 0;

    v8f acc[4] = {};
    const uint16_t* wrow = wc + (size_t)n0 * STATE_N;
    const uint16_t* arow = hb + (size_t)(m0 + lrow) * STATE_N + kh * 16;

#if HAVE_TDM
    if (wave == 0) {
        tdm_issue_2d(wrow, 0u, 64, STATE_N);
        wait_tensorcnt0();
    }
#else
    stage_slab_fallback(&smem[0], wrow, tid, STATE_N);
#endif
    __syncthreads();

    for (int ks = 0; ks < STATE_N / KSLAB; ++ks) {
        const int buf = ks & 1;
        if (ks < STATE_N / KSLAB - 1) {
#if HAVE_TDM
            if (wave == 0)
                tdm_issue_2d(wrow + (ks + 1) * KSLAB,
                             (uint32_t)((buf ^ 1) * SLAB_ELEMS * 2), 64, STATE_N);
#else
            stage_slab_fallback(&smem[(buf ^ 1) * SLAB_ELEMS],
                                wrow + (ks + 1) * KSLAB, tid, STATE_N);
#endif
        }

        v16bf af[4];
#pragma unroll
        for (int kk = 0; kk < 4; ++kk)
            af[kk] = *(const v16bf*)(arow + ks * KSLAB + kk * 32);

        const uint16_t* sw = &smem[buf * SLAB_ELEMS];
#pragma unroll
        for (int kk = 0; kk < 4; ++kk) {
#pragma unroll
            for (int j = 0; j < 4; ++j) {
                const int off = (j * 16 + lrow) * LDS_PITCH + kk * 32 + kh * 16;
                const v16bf bfrag = *(const v16bf*)(sw + off);
                acc[j] = wmma_bf16(af[kk], bfrag, acc[j]);
            }
        }

#if HAVE_TDM
        if (wave == 0) wait_tensorcnt0();
#endif
        __syncthreads();
    }

#pragma unroll
    for (int j = 0; j < 4; ++j) {
        const int col  = n0 + j * 16 + lrow;
        const float bs = bc[col];
        const float dp = Dp[col];
#pragma unroll
        for (int r = 0; r < 8; ++r) {
            const int rr = m0 + r + 8 * kh;
            const size_t idx = (size_t)rr * D_MODEL + col;
            y[idx] = acc[j][r] + bs + dp * x[idx];
        }
    }
}

// ================================================================ host launcher
extern "C" void kernel_launch(void* const* d_in, const int* in_sizes, int n_in,
                              void* d_out, int out_size, void* d_ws, size_t ws_size,
                              hipStream_t stream) {
    (void)in_sizes; (void)n_in; (void)out_size; (void)ws_size;
    const float* x  = (const float*)d_in[0];
    const float* Wa = (const float*)d_in[1];
    const float* ba = (const float*)d_in[2];
    const float* Wb = (const float*)d_in[3];
    const float* bb = (const float*)d_in[4];
    const float* Wc = (const float*)d_in[5];
    const float* bc = (const float*)d_in[6];
    const float* Dp = (const float*)d_in[7];
    const float* lw = (const float*)d_in[8];
    const float* lb = (const float*)d_in[9];
    float* y = (float*)d_out;

    const size_t MD = (size_t)M_TOTAL * D_MODEL;       // 16M elements
    const size_t WW = (size_t)STATE_N * D_MODEL;       // 1M elements
    char* p = (char*)d_ws;
    uint16_t* xnb  = (uint16_t*)p; p += MD * 2;        // 32 MB
    uint16_t* wab  = (uint16_t*)p; p += WW * 2;        //  2 MB
    uint16_t* wbb  = (uint16_t*)p; p += WW * 2;        //  2 MB
    uint16_t* wcb  = (uint16_t*)p; p += WW * 2;        //  2 MB
    uint16_t* alpb = (uint16_t*)p; p += MD * 2;        // 32 MB
    uint16_t* ub   = (uint16_t*)p; p += MD * 2;        // 32 MB
    uint16_t* hb   = (uint16_t*)p; p += MD * 2;        // 32 MB
    float* Pc    = (float*)p; p += (size_t)BATCH * N_CHUNK * STATE_N * 4;
    float* Sc    = (float*)p; p += (size_t)BATCH * N_CHUNK * STATE_N * 4;
    float* Hinit = (float*)p; p += (size_t)BATCH * N_CHUNK * STATE_N * 4;

    cvt_bf16_kernel<<<(int)(WW / 256), 256, 0, stream>>>(Wa, wab, (int)WW);
    cvt_bf16_kernel<<<(int)(WW / 256), 256, 0, stream>>>(Wb, wbb, (int)WW);
    cvt_bf16_kernel<<<(int)(WW / 256), 256, 0, stream>>>(Wc, wcb, (int)WW);

    ln_kernel<<<M_TOTAL / 8, 256, 0, stream>>>(x, lw, lb, xnb);

    {
        dim3 grid(STATE_N / 64, M_TOTAL / 128);
        gemm_alpha_u_kernel<<<grid, 256, 0, stream>>>(xnb, wab, wbb, ba, bb, alpb, ub);
    }
    {
        dim3 g1(STATE_N / 256, N_CHUNK, BATCH);
        scan_pass1_kernel<<<g1, 256, 0, stream>>>(alpb, ub, Pc, Sc);
        dim3 g2(STATE_N / 256, BATCH);
        scan_carry_kernel<<<g2, 256, 0, stream>>>(Pc, Sc, Hinit);
        scan_pass2_kernel<<<g1, 256, 0, stream>>>(alpb, ub, Hinit, hb);
    }
    {
        dim3 grid(D_MODEL / 64, M_TOTAL / 128);
        gemm_out_kernel<<<grid, 256, 0, stream>>>(hb, wcb, bc, Dp, x, y);
    }
}
